// net_gcn2_84524956385826
// MI455X (gfx1250) — matrix operand order repaired and checked by reading, not compiled, not run on hardware
//
#include <hip/hip_runtime.h>
#include <hip/hip_bf16.h>

typedef __attribute__((ext_vector_type(16))) _Float16 v16h;
typedef __attribute__((ext_vector_type(8)))  float    v8f;

#define WMMA_F16(a, b, c) \
  __builtin_amdgcn_wmma_f32_16x16x32_f16(false, (a), false, (b), (short)0, (c), false, false)

// Wave-local LDS producer->consumer fence (per-wave buffers; no cross-wave
// sharing so a full s_barrier is unnecessary). LDS ops are in-order (DScnt);
// the asm memory clobber stops compiler reordering, s_wait_dscnt drains.
__device__ __forceinline__ void lds_fence() {
  asm volatile("s_wait_dscnt 0" ::: "memory");
}

// Native f32 atomic add (global_atomic_add_f32, not a CAS loop):
// unsafeAtomicAdd carries the no-fine-grained-memory guarantee the backend
// needs to select the hardware fp atomic.
__device__ __forceinline__ void atomAddF(float* p, float v) {
  unsafeAtomicAdd(p, v);
}

// ---- WMMA fragment builders (layouts per CDNA5 ISA 7.12.2) ----
// A 16x32 f16 fragment from one f32 row (caller passes this lane's row ptr).
__device__ __forceinline__ v16h fragA_g32(const float* __restrict__ rowp, int k0, int lane) {
  v16h a;
  const int kb = (lane & 16) ? 8 : 0;
#pragma unroll
  for (int h = 0; h < 16; ++h) {
    const int v = h >> 1, p = h & 1;
    const int k = k0 + ((v & 4) ? 16 : 0) + kb + ((v & 3) << 1) + p;
    a[h] = (_Float16)rowp[k];
  }
  return a;
}
// A 16x32 f16 fragment from an f16 LDS row.
__device__ __forceinline__ v16h fragA_l16(const _Float16* __restrict__ rowp, int k0, int lane) {
  v16h a;
  const int kb = (lane & 16) ? 8 : 0;
#pragma unroll
  for (int h = 0; h < 16; ++h) {
    const int v = h >> 1, p = h & 1;
    const int k = k0 + ((v & 4) ? 16 : 0) + kb + ((v & 3) << 1) + p;
    a[h] = rowp[k];
  }
  return a;
}
// B 32x16 f16 fragment from row-major [K][ldn] f16 weights: N = lane%16, K = table order over half idx.
__device__ __forceinline__ v16h fragB_g16(const _Float16* __restrict__ w, int ldn,
                                          int k0, int n0, int lane) {
  v16h b;
  const int kb = (lane & 16) ? 8 : 0;
  const _Float16* p = w + n0 + (lane & 15);
#pragma unroll
  for (int h = 0; h < 16; ++h) {
    const int k = k0 + ((h & 8) ? 16 : 0) + kb + (h & 7);
    b[h] = p[(size_t)k * ldn];
  }
  return b;
}

// ---------------- small utility kernels ----------------
__global__ void k_f32tof16(const float* __restrict__ in, _Float16* __restrict__ out, int n) {
  int i = blockIdx.x * blockDim.x + threadIdx.x;
  if (i < n) out[i] = (_Float16)in[i];
}
__global__ void k_prelu16(const float* __restrict__ in, _Float16* __restrict__ out,
                          float scale, int n) {
  int i = blockIdx.x * blockDim.x + threadIdx.x;
  if (i < n) { float t = scale * in[i]; out[i] = (_Float16)(t > 0.f ? t : 0.f); }
}
__global__ void k_init(float* __restrict__ deg, float* __restrict__ stats, int N) {
  int i = blockIdx.x * blockDim.x + threadIdx.x;
  if (i < N) deg[i] = 1.0f;   // self-loop weight 1 folded into degree
  if (i < 4) stats[i] = 0.0f;
}

// ---------------- fused node MLP (all WMMA) ----------------
// Per wave: one 16-row node tile. Computes h1=relu(xW1+b1), h2=relu(h1W2+b2),
// logits=h2W3+b3 (stored f32), q=logits@P (stored f32), x0=relu(x lin0 + b).
__global__ __launch_bounds__(64) void k_mlp(
    const float* __restrict__ x,
    const float* __restrict__ b1, const float* __restrict__ b2,
    const float* __restrict__ b3, const float* __restrict__ l0b,
    const _Float16* __restrict__ w1h, const _Float16* __restrict__ w2h,
    const _Float16* __restrict__ w3h, const _Float16* __restrict__ Ph,
    const _Float16* __restrict__ l0h,
    float* __restrict__ lg, float* __restrict__ q, float* __restrict__ x0,
    int N) {
  __shared__ _Float16 h1s[2][16 * 512];
  __shared__ _Float16 h2s[2][16 * 64];
  __shared__ _Float16 lgs[2][16 * 64];
  const int wave = threadIdx.x >> 5;
  const int lane = threadIdx.x & 31;
  const int tile = blockIdx.x * 2 + wave;
  const int ntiles = (N + 15) >> 4;
  if (tile >= ntiles) return;            // wave-uniform; EXEC stays all-ones
  const long r0 = (long)tile << 4;
  const int m  = lane & 15;              // A-fragment row
  const int cn = lane & 15;              // C column
  const int mb = (lane >> 4) << 3;       // C row base
  long rA = r0 + m; if (rA >= N) rA = N - 1;

  v16h ax[4];
#pragma unroll
  for (int kc = 0; kc < 4; ++kc) ax[kc] = fragA_g32(x + rA * 128, kc * 32, lane);

  _Float16* h1w = h1s[wave];
  _Float16* h2w = h2s[wave];
  _Float16* lgw = lgs[wave];

  // h1 = relu(x @ W1 + b1)  [16,512]
  for (int nt = 0; nt < 32; ++nt) {
    const int n0 = nt << 4;
    v8f c = {};
#pragma unroll
    for (int kc = 0; kc < 4; ++kc)
      c = WMMA_F16(ax[kc], fragB_g16(w1h, 512, kc * 32, n0, lane), c);
    const float bias = b1[n0 + cn];
#pragma unroll
    for (int v = 0; v < 8; ++v) {
      float t = c[v] + bias;
      h1w[(mb + v) * 512 + n0 + cn] = (_Float16)(t > 0.f ? t : 0.f);
    }
  }
  lds_fence();

  // h2 = relu(h1 @ W2 + b2)  [16,64]
  for (int nt = 0; nt < 4; ++nt) {
    const int n0 = nt << 4;
    v8f c = {};
#pragma unroll
    for (int kc = 0; kc < 16; ++kc)
      c = WMMA_F16(fragA_l16(h1w + m * 512, kc * 32, lane),
                   fragB_g16(w2h, 64, kc * 32, n0, lane), c);
    const float bias = b2[n0 + cn];
#pragma unroll
    for (int v = 0; v < 8; ++v) {
      float t = c[v] + bias;
      h2w[(mb + v) * 64 + n0 + cn] = (_Float16)(t > 0.f ? t : 0.f);
    }
  }
  lds_fence();

  // logits = h2 @ W3 + b3  [16,64]
  for (int nt = 0; nt < 4; ++nt) {
    const int n0 = nt << 4;
    v8f c = {};
#pragma unroll
    for (int kc = 0; kc < 2; ++kc)
      c = WMMA_F16(fragA_l16(h2w + m * 64, kc * 32, lane),
                   fragB_g16(w3h, 64, kc * 32, n0, lane), c);
    const float bias = b3[n0 + cn];
#pragma unroll
    for (int v = 0; v < 8; ++v) {
      float t = c[v] + bias;
      const long row = r0 + mb + v;
      if (row < N) lg[row * 64 + n0 + cn] = t;
      lgw[(mb + v) * 64 + n0 + cn] = (_Float16)t;
    }
  }
  lds_fence();

  // q = logits @ P  [16,64]   (P = relu(SCALE*parsing[0]))
  for (int nt = 0; nt < 4; ++nt) {
    const int n0 = nt << 4;
    v8f c = {};
#pragma unroll
    for (int kc = 0; kc < 2; ++kc)
      c = WMMA_F16(fragA_l16(lgw + m * 64, kc * 32, lane),
                   fragB_g16(Ph, 64, kc * 32, n0, lane), c);
#pragma unroll
    for (int v = 0; v < 8; ++v) {
      const long row = r0 + mb + v;
      if (row < N) q[row * 64 + n0 + cn] = c[v];
    }
  }

  // x0 = relu(x @ lin0_w + lin0_b)  [16,128]  (reuses ax fragments)
  for (int nt = 0; nt < 8; ++nt) {
    const int n0 = nt << 4;
    v8f c = {};
#pragma unroll
    for (int kc = 0; kc < 4; ++kc)
      c = WMMA_F16(ax[kc], fragB_g16(l0h, 128, kc * 32, n0, lane), c);
    const float bias = l0b[n0 + cn];
#pragma unroll
    for (int v = 0; v < 8; ++v) {
      float t = c[v] + bias;
      const long row = r0 + mb + v;
      if (row < N) x0[row * 128 + n0 + cn] = (t > 0.f ? t : 0.f);
    }
  }
}

// ---------------- edge weight: ew[e] = dot(logits[row], q[col]) + stats ----
__global__ __launch_bounds__(256) void k_edgedot(
    const int* __restrict__ ei, const float* __restrict__ lg,
    const float* __restrict__ q, float* __restrict__ ew,
    float* __restrict__ stats, int E) {
  const long e = (long)blockIdx.x * 256 + threadIdx.x;
  float s = 0.f;
  if (e < E) {
    const float4* a = (const float4*)(lg + (long)ei[e] * 64);
    const float4* b = (const float4*)(q + (long)ei[E + e] * 64);
#pragma unroll
    for (int k = 0; k < 16; ++k) {
      float4 fa = a[k], fb = b[k];
      s += fa.x * fb.x + fa.y * fb.y + fa.z * fb.z + fa.w * fb.w;
    }
    ew[e] = s;
  }
  float ss = s * s;
#pragma unroll
  for (int off = 16; off > 0; off >>= 1) {
    s  += __shfl_xor(s, off, 32);
    ss += __shfl_xor(ss, off, 32);
  }
  if ((threadIdx.x & 31) == 0) { atomAddF(&stats[0], s); atomAddF(&stats[1], ss); }
}

__global__ void k_stats(float* __restrict__ stats, float invE, float invEm1, float Ef) {
  if (threadIdx.x == 0 && blockIdx.x == 0) {
    const float mean = stats[0] * invE;
    const float var  = (stats[1] - Ef * mean * mean) * invEm1;  // unbiased (ddof=1)
    stats[2] = mean;
    stats[3] = sqrtf(1e-4f / var);
  }
}

__global__ void k_normdeg(const int* __restrict__ ei, float* __restrict__ ew,
                          const float* __restrict__ stats, float* __restrict__ deg, int E) {
  const long e = (long)blockIdx.x * blockDim.x + threadIdx.x;
  if (e < E) {
    const float w = (ew[e] - stats[2]) * stats[3] + 1.0f;
    ew[e] = w;
    atomAddF(&deg[ei[E + e]], w);   // segment_sum over col
  }
}

__global__ void k_dinv(float* __restrict__ deg, int N) {
  const int i = blockIdx.x * blockDim.x + threadIdx.x;
  if (i < N) { const float d = deg[i]; deg[i] = d > 0.f ? rsqrtf(d) : 0.f; }
}

// agg initialized with the self-loop contribution: dinv[i]^2 * h[i]
__global__ void k_agginit(const float* __restrict__ dinv, const float* __restrict__ h,
                          float* __restrict__ agg, long total) {
  const long idx = (long)blockIdx.x * blockDim.x + threadIdx.x;
  if (idx < total) {
    const float di = dinv[idx >> 7];
    agg[idx] = di * di * h[idx];
  }
}

// one wave per edge: agg[col] += dinv[row]*ew*dinv[col] * h[row]   (128 floats)
__global__ __launch_bounds__(256) void k_scatter(
    const int* __restrict__ ei, const float* __restrict__ ew,
    const float* __restrict__ dinv, const float* __restrict__ h,
    float* __restrict__ agg, int E) {
  const int wave = threadIdx.x >> 5, lane = threadIdx.x & 31;
  const long e = (long)blockIdx.x * 8 + wave;
  if (e >= E) return;
  const int r = ei[e], c = ei[E + e];
  const float nrm = dinv[r] * ew[e] * dinv[c];
  const float4 hv = ((const float4*)(h + (long)r * 128))[lane];
  float* ac = agg + (long)c * 128 + lane * 4;
  atomAddF(ac + 0, nrm * hv.x);
  atomAddF(ac + 1, nrm * hv.y);
  atomAddF(ac + 2, nrm * hv.z);
  atomAddF(ac + 3, nrm * hv.w);
}

// h_out = relu((1-beta)*agg + beta*(agg @ conv_w1[ln]))   (WMMA)
__global__ __launch_bounds__(64) void k_combine(
    const float* __restrict__ agg, const _Float16* __restrict__ wh,
    float* __restrict__ hout, float beta, int N) {
  const int wave = threadIdx.x >> 5, lane = threadIdx.x & 31;
  const int tile = blockIdx.x * 2 + wave;
  const int ntiles = (N + 15) >> 4;
  if (tile >= ntiles) return;
  const long r0 = (long)tile << 4;
  const int m = lane & 15, cn = lane & 15, mb = (lane >> 4) << 3;
  long rA = r0 + m; if (rA >= N) rA = N - 1;
  v16h a[4];
#pragma unroll
  for (int kc = 0; kc < 4; ++kc) a[kc] = fragA_g32(agg + rA * 128, kc * 32, lane);
  const float omb = 1.0f - beta;
  for (int nt = 0; nt < 8; ++nt) {
    const int n0 = nt << 4;
    v8f c = {};
#pragma unroll
    for (int kc = 0; kc < 4; ++kc)
      c = WMMA_F16(a[kc], fragB_g16(wh, 128, kc * 32, n0, lane), c);
#pragma unroll
    for (int v = 0; v < 8; ++v) {
      const long row = r0 + mb + v;
      if (row < N) {
        const float t = omb * agg[row * 128 + n0 + cn] + beta * c[v];
        hout[row * 128 + n0 + cn] = t > 0.f ? t : 0.f;
      }
    }
  }
}

// out = h @ lin1_w + lin1_b  [N,64]   (WMMA)
__global__ __launch_bounds__(64) void k_final(
    const float* __restrict__ h, const _Float16* __restrict__ wh,
    const float* __restrict__ bias, float* __restrict__ out, int N) {
  const int wave = threadIdx.x >> 5, lane = threadIdx.x & 31;
  const int tile = blockIdx.x * 2 + wave;
  const int ntiles = (N + 15) >> 4;
  if (tile >= ntiles) return;
  const long r0 = (long)tile << 4;
  const int m = lane & 15, cn = lane & 15, mb = (lane >> 4) << 3;
  long rA = r0 + m; if (rA >= N) rA = N - 1;
  v16h a[4];
#pragma unroll
  for (int kc = 0; kc < 4; ++kc) a[kc] = fragA_g32(h + rA * 128, kc * 32, lane);
  for (int nt = 0; nt < 4; ++nt) {
    const int n0 = nt << 4;
    v8f c = {};
#pragma unroll
    for (int kc = 0; kc < 4; ++kc)
      c = WMMA_F16(a[kc], fragB_g16(wh, 64, kc * 32, n0, lane), c);
    const float b = bias[n0 + cn];
#pragma unroll
    for (int v = 0; v < 8; ++v) {
      const long row = r0 + mb + v;
      if (row < N) out[row * 64 + n0 + cn] = c[v] + b;
    }
  }
}

extern "C" void kernel_launch(void* const* d_in, const int* in_sizes, int n_in,
                              void* d_out, int out_size, void* d_ws, size_t ws_size,
                              hipStream_t stream) {
  const float* x       = (const float*)d_in[0];
  const int*   ei      = (const int*)d_in[1];
  const float* w1      = (const float*)d_in[2];
  const float* b1      = (const float*)d_in[3];
  const float* w2      = (const float*)d_in[4];
  const float* b2      = (const float*)d_in[5];
  const float* w3      = (const float*)d_in[6];
  const float* b3      = (const float*)d_in[7];
  const float* parsing = (const float*)d_in[8];
  const float* l0w     = (const float*)d_in[9];
  const float* l0b     = (const float*)d_in[10];
  const float* l1w     = (const float*)d_in[11];
  const float* l1b     = (const float*)d_in[12];
  const float* cw1     = (const float*)d_in[13];
  // d_in[14] (conv_w2) is dead: alpha == 0 makes its input exactly zero.

  const int N = in_sizes[0] / 128;
  const int E = in_sizes[1] / 2;

  char* base = (char*)d_ws;
  size_t off = 0;
  auto alloc = [&](size_t bytes) -> char* {
    off = (off + 255) & ~(size_t)255;
    char* p = base + off;
    off += bytes;
    return p;
  };
  float*    lg    = (float*)alloc((size_t)N * 64 * 4);
  float*    q     = (float*)alloc((size_t)N * 64 * 4);
  float*    buf0  = (float*)alloc((size_t)N * 128 * 4);   // h / x0
  float*    buf1  = (float*)alloc((size_t)N * 128 * 4);   // agg
  float*    ew    = (float*)alloc((size_t)E * 4);
  float*    deg   = (float*)alloc((size_t)N * 4);         // deg -> dinv in place
  float*    stats = (float*)alloc(256);
  _Float16* w1h   = (_Float16*)alloc((size_t)128 * 512 * 2);
  _Float16* w2h   = (_Float16*)alloc((size_t)512 * 64 * 2);
  _Float16* w3h   = (_Float16*)alloc((size_t)64 * 64 * 2);
  _Float16* Ph    = (_Float16*)alloc((size_t)64 * 64 * 2);
  _Float16* l0h   = (_Float16*)alloc((size_t)128 * 128 * 2);
  _Float16* l1h   = (_Float16*)alloc((size_t)128 * 64 * 2);
  _Float16* cw1h  = (_Float16*)alloc((size_t)2 * 128 * 128 * 2);

  auto cvt = [&](const float* in, _Float16* o, int n) {
    k_f32tof16<<<(n + 255) / 256, 256, 0, stream>>>(in, o, n);
  };
  cvt(w1, w1h, 128 * 512);
  cvt(w2, w2h, 512 * 64);
  cvt(w3, w3h, 64 * 64);
  cvt(l0w, l0h, 128 * 128);
  cvt(l1w, l1h, 128 * 64);
  cvt(cw1, cw1h, 2 * 128 * 128);
  k_prelu16<<<(64 * 64 + 255) / 256, 256, 0, stream>>>(parsing, Ph, 2.0f, 64 * 64);
  k_init<<<(N + 255) / 256, 256, 0, stream>>>(deg, stats, N);

  const int ntiles = (N + 15) / 16;
  const int gt = (ntiles + 1) / 2;   // 2 waves (tiles) per 64-thread block
  k_mlp<<<gt, 64, 0, stream>>>(x, b1, b2, b3, l0b, w1h, w2h, w3h, Ph, l0h,
                               lg, q, buf0, N);

  k_edgedot<<<(E + 255) / 256, 256, 0, stream>>>(ei, lg, q, ew, stats, E);
  k_stats<<<1, 32, 0, stream>>>(stats, 1.0f / (float)E, 1.0f / (float)(E - 1), (float)E);
  k_normdeg<<<(E + 255) / 256, 256, 0, stream>>>(ei, ew, stats, deg, E);
  k_dinv<<<(N + 255) / 256, 256, 0, stream>>>(deg, N);

  const long tot = (long)N * 128;
  const float beta[2] = {0.6931471805599453f, 0.4054651081081644f};  // log(1/(l+1)+1)
  for (int ln = 0; ln < 2; ++ln) {
    k_agginit<<<(int)((tot + 255) / 256), 256, 0, stream>>>(deg, buf0, buf1, tot);
    k_scatter<<<(E + 7) / 8, 256, 0, stream>>>(ei, ew, deg, buf0, buf1, E);
    k_combine<<<gt, 64, 0, stream>>>(buf1, cw1h + (size_t)ln * 128 * 128, buf0, beta[ln], N);
  }
  k_final<<<gt, 64, 0, stream>>>(buf0, l1h, l1b, (float*)d_out, N);
}
